// Qwen3Attention_32091995635827
// MI455X (gfx1250) — compile-verified, matrix operand order checked by reading, not run on hardware
//
#include <hip/hip_runtime.h>
#include <hip/hip_bf16.h>
#include <cstdint>

// ---------------------------------------------------------------------------
// Problem constants (from reference)
// ---------------------------------------------------------------------------
#define B_  4
#define S_  1024
#define T_  (B_ * S_)          // 4096
#define HQ_ 16
#define HKV_ 8
#define D_  128
#define HID_ 2048
#define NUM_SLOTS_ (512 * 16)  // 8192
#define CACHE_HALF_ ((size_t)NUM_SLOTS_ * HKV_ * D_)   // 8,388,608 floats
#define CACHE_ELEMS_ (2 * CACHE_HALF_)                 // 16,777,216 floats

typedef __attribute__((ext_vector_type(16))) __bf16 v16bf;
typedef __attribute__((ext_vector_type(8)))  float  v8f;

union AFrag { v16bf v; __bf16 e[16]; uint4 q[2]; };
union BFrag { v16bf v; __bf16 e[16]; uint4 q[2]; };

__device__ __forceinline__ v8f wmma_bf16(v16bf a, v16bf b, v8f c) {
  return __builtin_amdgcn_wmma_f32_16x16x32_bf16(false, a, false, b,
                                                 (short)0, c, false, false);
}

// ---------------------------------------------------------------------------
// gfx1250 async global->LDS copy (16B per lane), ASYNCcnt-tracked.
// INST_OFFSET applies to both the LDS destination and the global source.
// LDS address = low 32 bits of the generic pointer (ISA 10.2 aperture rule).
// ---------------------------------------------------------------------------
template <int OFF>
__device__ __forceinline__ void async_b128(void* lds_ptr, const void* gptr) {
  unsigned lds = (unsigned)(size_t)lds_ptr;
  unsigned long long ga = (unsigned long long)(size_t)gptr;
  asm volatile("global_load_async_to_lds_b128 %0, %1, off offset:%2"
               :: "v"(lds), "v"(ga), "n"(OFF) : "memory");
}
__device__ __forceinline__ void wait_async_lds() {
  asm volatile("s_wait_asynccnt 0" ::: "memory");
}

// ---------------------------------------------------------------------------
// f32 -> bf16 conversion (grid-stride)
// ---------------------------------------------------------------------------
__global__ void f32_to_bf16_kernel(const float* __restrict__ in,
                                   __bf16* __restrict__ out, int n) {
  int i = blockIdx.x * blockDim.x + threadIdx.x;
  int stride = gridDim.x * blockDim.x;
  for (; i < n; i += stride) out[i] = (__bf16)in[i];
}

// ---------------------------------------------------------------------------
// Tiled transpose + convert: in (K, N) f32 row-major -> out (N, K) bf16.
// Grid: (K/32, N/32), 256 threads (32x8).
// ---------------------------------------------------------------------------
__global__ __launch_bounds__(256) void transpose_f32_to_bf16_kernel(
    const float* __restrict__ in, __bf16* __restrict__ out, int K, int N) {
  __shared__ float tile[32][33];
  const int k0 = blockIdx.x * 32, n0 = blockIdx.y * 32;
  const int tx = threadIdx.x & 31, ty = threadIdx.x >> 5;
#pragma unroll
  for (int i = 0; i < 32; i += 8)
    tile[ty + i][tx] = in[(size_t)(k0 + ty + i) * N + n0 + tx];
  __syncthreads();
#pragma unroll
  for (int i = 0; i < 32; i += 8)
    out[(size_t)(n0 + ty + i) * K + k0 + tx] = (__bf16)tile[tx][ty + i];
}

// ---------------------------------------------------------------------------
// WMMA GEMM: C(f32, MxN) = A(bf16, MxK row-major) * BT(bf16, NxK row-major).
// N, K compile-time. Block = 256 threads (8 waves), tile 128x128x32; each
// wave computes 64x32 via 4x2 accumulators. Double-buffered async-LDS
// staging: prefetch tile k+1 while WMMAs consume tile k; 1 barrier/iter.
// ---------------------------------------------------------------------------
template <int N, int K>
__global__ __launch_bounds__(256) void gemm_bf16_wmma_kernel(
    const __bf16* __restrict__ A, const __bf16* __restrict__ BT,
    float* __restrict__ C, int M) {
  constexpr int BM = 128, BN = 128, BK = 32;
  constexpr int APAD = 40;   // elements; 80B rows keep 16B alignment
  constexpr int BPAD = 40;
  __shared__ __align__(64) __bf16 As[2][BM * APAD];
  __shared__ __align__(64) __bf16 Bs[2][BN * BPAD];   // [n][k]

  const int tid  = threadIdx.x;
  const int lane = tid & 31;
  const int wave = tid >> 5;
  const int half = lane >> 4;
  const int lr   = lane & 15;
  const int wm   = wave >> 2;   // 0..1
  const int wn   = wave & 3;    // 0..3
  const int m0   = blockIdx.y * BM;
  const int n0   = blockIdx.x * BN;

  v8f acc[4][2];
#pragma unroll
  for (int mt = 0; mt < 4; mt++)
#pragma unroll
    for (int nt = 0; nt < 2; nt++) acc[mt][nt] = (v8f){};

  const int a_r = tid >> 1;           // 0..127
  const int a_c = (tid & 1) * 16;     // 0 or 16
  const __bf16* a_src = A + (unsigned)(m0 + a_r) * K + a_c;
  const __bf16* b_src = BT + (unsigned)(n0 + a_r) * K + a_c;  // 128x32 shape
  const int lds_off = a_r * APAD + a_c;

  auto stage = [&](int kk, int buf) {
    async_b128<0>(&As[buf][lds_off], a_src + kk);
    async_b128<16>(&As[buf][lds_off], a_src + kk);
    async_b128<0>(&Bs[buf][lds_off], b_src + kk);
    async_b128<16>(&Bs[buf][lds_off], b_src + kk);
  };

  stage(0, 0);  // prologue prefetch
  for (int kk = 0; kk < K; kk += BK) {
    const int buf = (kk >> 5) & 1;
    wait_async_lds();
    __syncthreads();                       // tile kk visible block-wide
    if (kk + BK < K) stage(kk + BK, buf ^ 1);

    BFrag bf[2];
#pragma unroll
    for (int nt = 0; nt < 2; nt++) {
      const int n = wn * 32 + nt * 16 + lr;
      bf[nt].q[0] = *(const uint4*)(&Bs[buf][n * BPAD + half * 8]);
      bf[nt].q[1] = *(const uint4*)(&Bs[buf][n * BPAD + 16 + half * 8]);
    }
    AFrag af[4];
#pragma unroll
    for (int mt = 0; mt < 4; mt++) {
      const int r = wm * 64 + mt * 16 + lr;
      af[mt].q[0] = *(const uint4*)(&As[buf][r * APAD + half * 8]);
      af[mt].q[1] = *(const uint4*)(&As[buf][r * APAD + 16 + half * 8]);
    }
#pragma unroll
    for (int mt = 0; mt < 4; mt++)
#pragma unroll
      for (int nt = 0; nt < 2; nt++)
        acc[mt][nt] = wmma_bf16(af[mt].v, bf[nt].v, acc[mt][nt]);
  }

  // Store: C/D layout -> m = j + 8*half, n = lr (32-bit address math, N const)
  const unsigned colbase = (unsigned)(n0 + wn * 32 + lr);
#pragma unroll
  for (int mt = 0; mt < 4; mt++)
#pragma unroll
    for (int nt = 0; nt < 2; nt++) {
      const unsigned base =
          (unsigned)(m0 + wm * 64 + mt * 16 + 8 * half) * N + colbase + nt * 16;
#pragma unroll
      for (int j = 0; j < 8; j++) C[base + (unsigned)j * N] = acc[mt][nt][j];
    }
}

// ---------------------------------------------------------------------------
// Fused RMSNorm + RoPE on (T, n_heads, 128) f32, in place, plus a bf16 copy
// scaled by out_scale (1/sqrt(D) folded in for Q; 1.0 for K).
// ---------------------------------------------------------------------------
__global__ __launch_bounds__(128) void rmsnorm_rope_kernel(
    float* __restrict__ data, __bf16* __restrict__ out16,
    const float* __restrict__ scale, const int* __restrict__ positions,
    int n_heads, float out_scale) {
  const int bid = blockIdx.x;           // t * n_heads + h
  const int t   = bid / n_heads;
  const int d   = threadIdx.x;          // 0..127
  float v = data[(size_t)bid * 128 + d];

  float ss = v * v;
#pragma unroll
  for (int o = 16; o > 0; o >>= 1) ss += __shfl_xor(ss, o);
  __shared__ float red[4];
  if ((threadIdx.x & 31) == 0) red[threadIdx.x >> 5] = ss;
  __syncthreads();
  const float tot = red[0] + red[1] + red[2] + red[3];
  const float inv = rsqrtf(tot * (1.0f / 128.0f) + 1e-6f);
  const float xn  = v * inv * scale[d];

  const float pos  = (float)positions[t];
  const int   pair = d >> 1;
  const float inv_freq = __expf(-(float)(2 * pair) * (13.815510557964274f / 128.0f));
  const float fr = pos * inv_freq;
  float s, c;
  __sincosf(fr, &s, &c);
  const float partner = __shfl_xor(xn, 1);
  const float out = ((d & 1) == 0) ? (xn * c - partner * s)
                                   : (partner * s + xn * c);
  data[(size_t)bid * 128 + d]  = out;
  out16[(size_t)bid * 128 + d] = (__bf16)(out * out_scale);
}

// ---------------------------------------------------------------------------
// V: f32 (T, HKV, D) -> bf16 (B, HKV, D, S) transposed for async V staging.
// ---------------------------------------------------------------------------
__global__ void v_transpose_bf16_kernel(const float* __restrict__ vraw,
                                        __bf16* __restrict__ vt) {
  const int i   = blockIdx.x * blockDim.x + threadIdx.x;  // < T*HKV*D
  const int t   = i >> 10;
  const int rem = i & 1023;
  const int h   = rem >> 7;
  const int d   = rem & 127;
  const int b   = t >> 10;           // S = 1024
  const int s   = t & 1023;
  vt[(((size_t)(b * HKV_ + h) * D_ + d) << 10) + s] = (__bf16)vraw[i];
}

// ---------------------------------------------------------------------------
// Scatter rope'd K and raw V into the (already copied) output cache.
// ---------------------------------------------------------------------------
__global__ void scatter_kv_kernel(const float* __restrict__ kraw,
                                  const float* __restrict__ vraw,
                                  const int* __restrict__ slot_map,
                                  float* __restrict__ out_cache) {
  const int i = blockIdx.x * blockDim.x + threadIdx.x;  // < T*HKV*D
  const int t   = i >> 10;
  const int rem = i & 1023;
  const int s   = slot_map[t];
  out_cache[(size_t)s * 1024 + rem]               = kraw[i];
  out_cache[CACHE_HALF_ + (size_t)s * 1024 + rem] = vraw[i];
}

// ---------------------------------------------------------------------------
// Online-softmax tile update; MASKED=false is the fast path for key blocks
// fully below the causal diagonal (wave-uniform branch, EXEC stays all-1s).
// ---------------------------------------------------------------------------
template <bool MASKED>
__device__ __forceinline__ void softmax_update(
    const v8f& s0, const v8f& s1, v8f* acc, float* rm, float* rl,
    __bf16* pb, int qr0, int k0, int lr, int half) {
  constexpr int PPAD = 40;
#pragma unroll
  for (int j = 0; j < 8; j++) {
    const int m = j + 8 * half;
    float e0, e1;
    if (MASKED) {
      const int sq = qr0 + m;
      e0 = ((k0 + lr) <= sq)      ? s0[j] : -3.0e38f;
      e1 = ((k0 + 16 + lr) <= sq) ? s1[j] : -3.0e38f;
    } else {
      e0 = s0[j];
      e1 = s1[j];
    }
    float tmax = fmaxf(e0, e1);
#pragma unroll
    for (int o = 1; o < 16; o <<= 1) tmax = fmaxf(tmax, __shfl_xor(tmax, o));
    const float mn = fmaxf(rm[j], tmax);
    const float p0 = __expf(e0 - mn);
    const float p1 = __expf(e1 - mn);
    float rs = p0 + p1;
#pragma unroll
    for (int o = 1; o < 16; o <<= 1) rs += __shfl_xor(rs, o);
    const float sc = __expf(rm[j] - mn);
    rl[j] = rl[j] * sc + rs;
    rm[j] = mn;
#pragma unroll
    for (int nt = 0; nt < 8; nt++) acc[nt][j] *= sc;
    pb[m * PPAD + lr]      = (__bf16)p0;
    pb[m * PPAD + 16 + lr] = (__bf16)p1;
  }
}

// ---------------------------------------------------------------------------
// Flash-style causal GQA attention. Grid: (S/64, HQ, B); 128 threads = 4
// waves, 16 q-rows each. K and transposed-V staged with double-buffered
// async-LDS copies; QK^T and PV on v_wmma_f32_16x16x32_bf16.
// ---------------------------------------------------------------------------
__global__ __launch_bounds__(128) void attention_kernel(
    const __bf16* __restrict__ qb, const __bf16* __restrict__ kb,
    const __bf16* __restrict__ vt, __bf16* __restrict__ og) {
  constexpr int KPAD = 136;  // 272B rows (16B aligned)
  constexpr int VPAD = 40;   // 80B rows
  constexpr int PPAD = 40;
  __shared__ __align__(64) __bf16 Ks[2][32 * KPAD];    // [key][d]
  __shared__ __align__(64) __bf16 Vs[2][128 * VPAD];   // [d][key]
  __shared__ __align__(64) __bf16 Pb[4][16 * PPAD];    // per-wave P scratch

  const int b = blockIdx.z, h = blockIdx.y, qb_i = blockIdx.x;
  const int kvh  = h >> 1;
  const int tid  = threadIdx.x;
  const int lane = tid & 31;
  const int wave = tid >> 5;
  const int half = lane >> 4;
  const int lr   = lane & 15;
  const int qr0  = qb_i * 64 + wave * 16;

  // Async staging addresses (fixed per thread; global side advances with k0)
  const int key = tid >> 2;              // 0..31
  const int d0  = (tid & 3) * 32;
  const int ks_off = key * KPAD + d0;
  const int vs_off = tid * VPAD;         // d = tid
  const __bf16* ks_src0 =
      kb + ((size_t)(b * S_ + key) * HKV_ + kvh) * D_ + d0;   // + k0*HKV*D
  const __bf16* vs_src0 =
      vt + (((size_t)(b * HKV_ + kvh) * D_ + tid) << 10);     // + k0

  auto stage = [&](int k0, int buf) {
    const __bf16* ks_src = ks_src0 + (size_t)k0 * HKV_ * D_;
    async_b128<0>(&Ks[buf][ks_off], ks_src);
    async_b128<16>(&Ks[buf][ks_off], ks_src);
    async_b128<32>(&Ks[buf][ks_off], ks_src);
    async_b128<48>(&Ks[buf][ks_off], ks_src);
    const __bf16* vs_src = vs_src0 + k0;
    async_b128<0>(&Vs[buf][vs_off], vs_src);
    async_b128<16>(&Vs[buf][vs_off], vs_src);
    async_b128<32>(&Vs[buf][vs_off], vs_src);
    async_b128<48>(&Vs[buf][vs_off], vs_src);
  };

  stage(0, 0);  // prologue prefetch (overlaps with Q fragment loads below)

  // Q fragments: A-layout; 8-element runs are 16B-contiguous in global bf16.
  AFrag qf[4];
  {
    const __bf16* qp = qb + ((size_t)(b * S_ + qr0 + lr) * HQ_ + h) * D_;
#pragma unroll
    for (int kc = 0; kc < 4; kc++) {
      qf[kc].q[0] = *(const uint4*)(qp + kc * 32 + half * 8);
      qf[kc].q[1] = *(const uint4*)(qp + kc * 32 + 16 + half * 8);
    }
  }

  v8f acc[8];
#pragma unroll
  for (int nt = 0; nt < 8; nt++) acc[nt] = (v8f){};
  float rm[8], rl[8];
#pragma unroll
  for (int j = 0; j < 8; j++) { rm[j] = -3.0e38f; rl[j] = 0.0f; }

  const int kmax = qb_i * 64 + 64;
  for (int k0 = 0; k0 < kmax; k0 += 32) {
    const int buf = (k0 >> 5) & 1;
    wait_async_lds();
    __syncthreads();                       // K/V tile k0 visible block-wide
    if (k0 + 32 < kmax) stage(k0 + 32, buf ^ 1);

    // Scores: two 16x16 key tiles, K-dim = D in 4 chunks of 32.
    v8f s0 = (v8f){}, s1 = (v8f){};
#pragma unroll
    for (int kc = 0; kc < 4; kc++) {
      BFrag kf0, kf1;
      kf0.q[0] = *(const uint4*)(&Ks[buf][lr * KPAD + kc * 32 + half * 8]);
      kf0.q[1] = *(const uint4*)(&Ks[buf][lr * KPAD + kc * 32 + 16 + half * 8]);
      kf1.q[0] = *(const uint4*)(&Ks[buf][(16 + lr) * KPAD + kc * 32 + half * 8]);
      kf1.q[1] =
          *(const uint4*)(&Ks[buf][(16 + lr) * KPAD + kc * 32 + 16 + half * 8]);
      s0 = wmma_bf16(qf[kc].v, kf0.v, s0);
      s1 = wmma_bf16(qf[kc].v, kf1.v, s1);
    }

    // Softmax update; wave-uniform branch picks the unmasked fast path.
    __bf16* pb = &Pb[wave][0];
    if (k0 + 31 <= qr0)
      softmax_update<false>(s0, s1, acc, rm, rl, pb, qr0, k0, lr, half);
    else
      softmax_update<true>(s0, s1, acc, rm, rl, pb, qr0, k0, lr, half);

    // P (16x32) -> A-layout frag via per-wave LDS bounce (same-wave in-order).
    AFrag pf;
    pf.q[0] = *(const uint4*)(&pb[lr * PPAD + half * 8]);
    pf.q[1] = *(const uint4*)(&pb[lr * PPAD + 16 + half * 8]);

#pragma unroll
    for (int nt = 0; nt < 8; nt++) {
      BFrag vf;
      const int n = nt * 16 + lr;  // d index
      vf.q[0] = *(const uint4*)(&Vs[buf][n * VPAD + half * 8]);
      vf.q[1] = *(const uint4*)(&Vs[buf][n * VPAD + 16 + half * 8]);
      acc[nt] = wmma_bf16(pf.v, vf.v, acc[nt]);
    }
  }

  // Epilogue: divide by row sums, write bf16 (T, HQ*D).
#pragma unroll
  for (int nt = 0; nt < 8; nt++)
#pragma unroll
    for (int j = 0; j < 8; j++) {
      const int m = j + 8 * half;
      const int t = b * S_ + qr0 + m;
      const int d = nt * 16 + lr;
      og[((size_t)t * HQ_ + h) * D_ + d] = (__bf16)(acc[nt][j] / rl[j]);
    }
}

// ---------------------------------------------------------------------------
// Host-side launch
// ---------------------------------------------------------------------------
extern "C" void kernel_launch(void* const* d_in, const int* in_sizes, int n_in,
                              void* d_out, int out_size, void* d_ws,
                              size_t ws_size, hipStream_t stream) {
  const float* kv_in     = (const float*)d_in[0];
  const float* x         = (const float*)d_in[1];
  const float* Wq        = (const float*)d_in[2];
  const float* Wk        = (const float*)d_in[3];
  const float* Wv        = (const float*)d_in[4];
  const float* Wo        = (const float*)d_in[5];
  const float* q_scale   = (const float*)d_in[6];
  const float* k_scale   = (const float*)d_in[7];
  const int*   positions = (const int*)d_in[8];
  const int*   slot_map  = (const int*)d_in[9];

  float* out       = (float*)d_out;
  float* out_cache = out;
  float* out_o     = out + CACHE_ELEMS_;

  // Workspace layout (bytes)
  char* ws = (char*)d_ws;
  __bf16* xb   = (__bf16*)(ws + 0);              // T*HID bf16
  __bf16* wqT  = (__bf16*)(ws + 16777216);       // (2048,2048) bf16 (N,K)
  __bf16* wkT  = (__bf16*)(ws + 25165824);       // (1024,2048)
  __bf16* wvT  = (__bf16*)(ws + 29360128);       // (1024,2048)
  __bf16* woT  = (__bf16*)(ws + 33554432);       // (2048,2048)
  float*  qraw = (float*) (ws + 41943040);       // T*HQ*D f32 (reused as attnb)
  float*  kraw = (float*) (ws + 75497472);       // T*HKV*D f32
  float*  vraw = (float*) (ws + 92274688);       // T*HKV*D f32
  __bf16* qb16 = (__bf16*)(ws + 109051904);      // T*HQ*D bf16
  __bf16* kb16 = (__bf16*)(ws + 125829120);      // T*HKV*D bf16
  __bf16* vt16 = (__bf16*)(ws + 134217728);      // (B,HKV,D,S) bf16
  __bf16* attnb = (__bf16*)qraw;                 // alias: qraw dead after rope
  // workspace high-water: 142,606,336 bytes

  // 1) bf16 conversions; weights transposed to (N, K) for async staging.
  f32_to_bf16_kernel<<<2048, 256, 0, stream>>>(x, xb, T_ * HID_);
  transpose_f32_to_bf16_kernel<<<dim3(HID_ / 32, (HQ_ * D_) / 32), 256, 0,
                                 stream>>>(Wq, wqT, HID_, HQ_ * D_);
  transpose_f32_to_bf16_kernel<<<dim3(HID_ / 32, (HKV_ * D_) / 32), 256, 0,
                                 stream>>>(Wk, wkT, HID_, HKV_ * D_);
  transpose_f32_to_bf16_kernel<<<dim3(HID_ / 32, (HKV_ * D_) / 32), 256, 0,
                                 stream>>>(Wv, wvT, HID_, HKV_ * D_);
  transpose_f32_to_bf16_kernel<<<dim3((HQ_ * D_) / 32, HID_ / 32), 256, 0,
                                 stream>>>(Wo, woT, HQ_ * D_, HID_);

  // 2) QKV projections (WMMA GEMMs, double-buffered async-LDS staging)
  gemm_bf16_wmma_kernel<HQ_ * D_, HID_>
      <<<dim3((HQ_ * D_) / 128, T_ / 128), 256, 0, stream>>>(xb, wqT, qraw, T_);
  gemm_bf16_wmma_kernel<HKV_ * D_, HID_>
      <<<dim3((HKV_ * D_) / 128, T_ / 128), 256, 0, stream>>>(xb, wkT, kraw, T_);
  gemm_bf16_wmma_kernel<HKV_ * D_, HID_>
      <<<dim3((HKV_ * D_) / 128, T_ / 128), 256, 0, stream>>>(xb, wvT, vraw, T_);

  // 3) RMSNorm + RoPE; Q gets 1/sqrt(D) folded into its bf16 copy.
  rmsnorm_rope_kernel<<<T_ * HQ_, 128, 0, stream>>>(
      qraw, qb16, q_scale, positions, HQ_, 0.08838834764831845f);
  rmsnorm_rope_kernel<<<T_ * HKV_, 128, 0, stream>>>(
      kraw, kb16, k_scale, positions, HKV_, 1.0f);

  // 4) V transpose for async staging; KV cache copy + scatter.
  v_transpose_bf16_kernel<<<(T_ * HKV_ * D_) / 256, 256, 0, stream>>>(vraw, vt16);
  hipMemcpyAsync(out_cache, kv_in, CACHE_ELEMS_ * sizeof(float),
                 hipMemcpyDeviceToDevice, stream);
  scatter_kv_kernel<<<(T_ * HKV_ * D_) / 256, 256, 0, stream>>>(
      kraw, vraw, slot_map, out_cache);

  // 5) Causal GQA flash attention -> bf16 (T, HQ*D), aliased over qraw.
  attention_kernel<<<dim3(S_ / 64, HQ_, B_), 128, 0, stream>>>(
      qb16, kb16, vt16, attnb);

  // 6) Output projection
  gemm_bf16_wmma_kernel<HID_, HQ_ * D_>
      <<<dim3(HID_ / 128, T_ / 128), 256, 0, stream>>>(attnb, woT, out_o, T_);
}
